// LSTM_29102698398241
// MI455X (gfx1250) — matrix operand order repaired and checked by reading, not compiled, not run on hardware
//
#include <hip/hip_runtime.h>
#include <hip/hip_bf16.h>

#define D 1024
#define V 32000
#define T 2048

typedef float    v8f   __attribute__((ext_vector_type(8)));
typedef __bf16   v16bf __attribute__((ext_vector_type(16)));
typedef unsigned int v4u __attribute__((ext_vector_type(4)));
typedef int      v4i   __attribute__((ext_vector_type(4)));

// ---------------------------------------------------------------------------
// bf16 split helpers (RNE)
// ---------------------------------------------------------------------------
__device__ __forceinline__ unsigned short f2bf_rne(float x) {
    unsigned int u = __float_as_uint(x);
    u += 0x7FFFu + ((u >> 16) & 1u);
    return (unsigned short)(u >> 16);
}
__device__ __forceinline__ float bf2f(unsigned short b) {
    return __uint_as_float(((unsigned int)b) << 16);
}

// ---------------------------------------------------------------------------
// CDNA5 async memory->LDS copy (16B per lane), ASYNCcnt-tracked
// ---------------------------------------------------------------------------
__device__ __forceinline__ void async_copy16(const void* gsrc, void* ldst) {
#if __has_builtin(__builtin_amdgcn_global_load_async_to_lds_b128)
    __builtin_amdgcn_global_load_async_to_lds_b128(
        (__attribute__((address_space(1))) v4i*)gsrc,
        (__attribute__((address_space(3))) v4i*)ldst, 0, 0);
#else
    unsigned int loff =
        (unsigned int)(uintptr_t)(__attribute__((address_space(3))) void*)ldst;
    asm volatile("global_load_async_to_lds_b128 %0, %1, off"
                 :: "v"(loff), "v"((unsigned long long)(uintptr_t)gsrc)
                 : "memory");
#endif
}
__device__ __forceinline__ void wait_async0() {
#if __has_builtin(__builtin_amdgcn_s_wait_asynccnt)
    __builtin_amdgcn_s_wait_asynccnt(0);
#else
    asm volatile("s_wait_asynccnt 0" ::: "memory");
#endif
}

// ---------------------------------------------------------------------------
// Kernel 0: zero the cross-workgroup barrier counter (d_ws is poisoned).
// ---------------------------------------------------------------------------
__global__ void init_ws(unsigned int* ctr) { *ctr = 0u; }

// ---------------------------------------------------------------------------
// Kernel 1: LSTM recurrence, 32 cooperating workgroups x 128 threads.
// WG w owns hidden columns [32w, 32w+32). Thread = (gate g, column jl):
// one 1024-long dot per step. Gates combine via LDS; h_t broadcast via the
// (L2-resident) Hout row; steps separated by an agent-scope atomic barrier.
// ---------------------------------------------------------------------------
#define NWG 32
#define CPW 32    // columns per workgroup

__global__ __launch_bounds__(128)
void lstm_recur_mw(const int* __restrict__ X,
                   const float* __restrict__ Wxf, const float* __restrict__ Whf,
                   const float* __restrict__ Wxi, const float* __restrict__ Whi,
                   const float* __restrict__ Wxc, const float* __restrict__ Whc,
                   const float* __restrict__ Wxo, const float* __restrict__ Who,
                   const float* __restrict__ bf,  const float* __restrict__ bi,
                   const float* __restrict__ bc,  const float* __restrict__ bo,
                   float* __restrict__ Hout, unsigned int* __restrict__ ctr)
{
    __shared__ float hsh[D];
    __shared__ float gsh[4][CPW];

    const int w   = blockIdx.x;
    const int tid = threadIdx.x;
    const int g   = tid >> 5;       // gate 0..3 (wave-uniform)
    const int jl  = tid & 31;
    const int j   = w * CPW + jl;

    const float* WxA[4] = {Wxf, Wxi, Wxc, Wxo};
    const float* WhA[4] = {Whf, Whi, Whc, Who};
    const float* bA[4]  = {bf,  bi,  bc,  bo};
    const float* Wxg = WxA[g];
    const float* Whg = WhA[g];
    const float  bg  = bA[g][j];

    float c = 0.0f;

    for (int t = 0; t < T; ++t) {
        // stage h_{t-1} into LDS
        if (t == 0) {
            for (int k = tid; k < D; k += 128) hsh[k] = 0.0f;
        } else {
            const float* hp = &Hout[(size_t)(t - 1) * D];
            for (int k = tid; k < D; k += 128) hsh[k] = hp[k];
        }
        __syncthreads();

        const int id = X[t];
        float a = Wxg[(size_t)id * D + j] + bg;
        #pragma unroll 4
        for (int k = 0; k < D; ++k)
            a = fmaf(hsh[k], Whg[k * D + j], a);

        gsh[g][jl] = a;
        __syncthreads();

        if (tid < CPW) {
            const float f = 1.0f / (1.0f + __expf(-gsh[0][tid]));
            const float i = 1.0f / (1.0f + __expf(-gsh[1][tid]));
            const float gg = tanhf(gsh[2][tid]);
            const float o = 1.0f / (1.0f + __expf(-gsh[3][tid]));
            c = f * c + i * gg;
            Hout[(size_t)t * D + w * CPW + tid] = tanhf(c) * o;
        }
        __threadfence();
        __syncthreads();
        if (tid == 0) {
            __hip_atomic_fetch_add(ctr, 1u, __ATOMIC_RELEASE, __HIP_MEMORY_SCOPE_AGENT);
            const unsigned int target = (unsigned int)(NWG * (t + 1));
            while (__hip_atomic_load(ctr, __ATOMIC_ACQUIRE, __HIP_MEMORY_SCOPE_AGENT) < target) {}
        }
        __syncthreads();
    }
}

// ---------------------------------------------------------------------------
// Kernel 2a: one-time fp32 -> (hi,lo) bf16 split, 4 elements/thread.
// ---------------------------------------------------------------------------
__global__ __launch_bounds__(256)
void split_pair(const float* __restrict__ src,
                unsigned short* __restrict__ hi, unsigned short* __restrict__ lo)
{
    const size_t i = ((size_t)blockIdx.x * 256 + threadIdx.x) * 4;
    const float4 v = *(const float4*)&src[i];
    const unsigned short h0 = f2bf_rne(v.x), h1 = f2bf_rne(v.y),
                         h2 = f2bf_rne(v.z), h3 = f2bf_rne(v.w);
    const unsigned short l0 = f2bf_rne(v.x - bf2f(h0)), l1 = f2bf_rne(v.y - bf2f(h1)),
                         l2 = f2bf_rne(v.z - bf2f(h2)), l3 = f2bf_rne(v.w - bf2f(h3));
    *(unsigned int*)&hi[i]     = (unsigned int)h0 | ((unsigned int)h1 << 16);
    *(unsigned int*)&hi[i + 2] = (unsigned int)h2 | ((unsigned int)h3 << 16);
    *(unsigned int*)&lo[i]     = (unsigned int)l0 | ((unsigned int)l1 << 16);
    *(unsigned int*)&lo[i + 2] = (unsigned int)l2 | ((unsigned int)l3 << 16);
}

// ---------------------------------------------------------------------------
// WMMA fragment gathers (ISA 7.12.2 16-bit layouts), two ds_load_b128 each.
// ---------------------------------------------------------------------------
#define BT 64
#define BV 64
#define KB 32
#define KPAD 40            // shorts per LDS row: 80B, multiple of 16 -> conflict-free b128
#define BUFOFF (64 * KPAD) // per-buffer stride (elements), same for all tiles

__device__ __forceinline__ v16bf frag_a(const unsigned short* rowp, int kg) {
    union { v4u q[2]; v16bf v; } u;
    u.q[0] = *(const v4u*)(rowp + 8 * kg);
    u.q[1] = *(const v4u*)(rowp + 16 + 8 * kg);
    return u.v;
}
__device__ __forceinline__ v16bf frag_b(const unsigned short* rowp, int kg) {
    union { v4u q[2]; v16bf v; } u;
    u.q[0] = *(const v4u*)(rowp + 8 * kg);
    u.q[1] = *(const v4u*)(rowp + 16 + 8 * kg);
    return u.v;
}

// ---------------------------------------------------------------------------
// Kernel 2b (fast path): GEMM on pre-split bf16 operands. 64x64 block tile,
// 8 waves, two 16x16 tiles per wave, 3 WMMAs per operand pair per 32-K slab.
// Staging = pure async global->LDS b128 copies, double-buffered; all tile /
// buffer selection hoisted out of the K loop (buffer toggle = +BUFOFF).
// ---------------------------------------------------------------------------
__global__ __launch_bounds__(256)
void out_proj_pre(const unsigned short* __restrict__ Hhi, const unsigned short* __restrict__ Hlo,
                  const unsigned short* __restrict__ Whi, const unsigned short* __restrict__ Wlo,
                  const float* __restrict__ bias, float* __restrict__ Y)
{
    __shared__ __align__(16) unsigned short sHh[2][BT][KPAD];
    __shared__ __align__(16) unsigned short sHl[2][BT][KPAD];
    __shared__ __align__(16) unsigned short sWh[2][BV][KPAD];
    __shared__ __align__(16) unsigned short sWl[2][BV][KPAD];

    const int t0   = blockIdx.y * BT;
    const int v0   = blockIdx.x * BV;
    const int tid  = threadIdx.x;
    const int lane = tid & 31;
    const int wave = tid >> 5;
    const int tb   = (wave >> 2) * 32;
    const int tc   = wave & 3;
    const int m    = lane & 15;
    const int kg   = lane >> 4;

    // ---- staging assignment, computed once ----
    const int tile = tid >> 6;   // 0..3 : which LDS tile this thread stages
    const int rr   = tid & 63;   // row within tile
    const unsigned short* gbase;
    unsigned short* lbase;       // buffer-0 destination row
    switch (tile) {
        case 0:  gbase = Hhi + (size_t)(t0 + rr) * D; lbase = &sHh[0][rr][0]; break;
        case 1:  gbase = Hlo + (size_t)(t0 + rr) * D; lbase = &sHl[0][rr][0]; break;
        case 2:  gbase = Whi + (size_t)(v0 + rr) * D; lbase = &sWh[0][rr][0]; break;
        default: gbase = Wlo + (size_t)(v0 + rr) * D; lbase = &sWl[0][rr][0]; break;
    }

    // ---- fragment row pointers (buffer 0), computed once ----
    const unsigned short* pBh  = &sWh[0][tc * 16 + m][0];
    const unsigned short* pBl  = &sWl[0][tc * 16 + m][0];
    const unsigned short* pAh0 = &sHh[0][tb + m][0];
    const unsigned short* pAl0 = &sHl[0][tb + m][0];
    const unsigned short* pAh1 = &sHh[0][tb + 16 + m][0];
    const unsigned short* pAl1 = &sHl[0][tb + 16 + m][0];

    v8f acc0 = {};
    v8f acc1 = {};

    // prologue: fill buffer 0 with chunk 0
    #pragma unroll
    for (int s = 0; s < 4; ++s)
        async_copy16(gbase + s * 8, lbase + s * 8);
    wait_async0();
    __syncthreads();

    int cur = 0;
    for (int k0 = 0; k0 < D; k0 += KB) {
        const int nxt = cur ^ 1;
        if (k0 + KB < D) {
            const unsigned short* gsrc = gbase + k0 + KB;
            unsigned short* ldst = lbase + nxt * BUFOFF;
            #pragma unroll
            for (int s = 0; s < 4; ++s)
                async_copy16(gsrc + s * 8, ldst + s * 8);
        }

        const int bo = cur * BUFOFF;
        const v16bf bh  = frag_b(pBh  + bo, kg);
        const v16bf bl  = frag_b(pBl  + bo, kg);
        const v16bf ah0 = frag_a(pAh0 + bo, kg);
        const v16bf al0 = frag_a(pAl0 + bo, kg);
        const v16bf ah1 = frag_a(pAh1 + bo, kg);
        const v16bf al1 = frag_a(pAl1 + bo, kg);

        acc0 = __builtin_amdgcn_wmma_f32_16x16x32_bf16(false, ah0, false, bh, (short)0, acc0, false, false);
        acc0 = __builtin_amdgcn_wmma_f32_16x16x32_bf16(false, ah0, false, bl, (short)0, acc0, false, false);
        acc0 = __builtin_amdgcn_wmma_f32_16x16x32_bf16(false, al0, false, bh, (short)0, acc0, false, false);
        acc1 = __builtin_amdgcn_wmma_f32_16x16x32_bf16(false, ah1, false, bh, (short)0, acc1, false, false);
        acc1 = __builtin_amdgcn_wmma_f32_16x16x32_bf16(false, ah1, false, bl, (short)0, acc1, false, false);
        acc1 = __builtin_amdgcn_wmma_f32_16x16x32_bf16(false, al1, false, bh, (short)0, acc1, false, false);

        wait_async0();
        __syncthreads();
        cur = nxt;
    }

    const float bv = bias[v0 + tc * 16 + m];
    #pragma unroll
    for (int r = 0; r < 8; ++r) {
        const int M = r + 8 * kg;
        Y[(size_t)(t0 + tb + M) * V      + (v0 + tc * 16 + m)] = acc0[r] + bv;
        Y[(size_t)(t0 + tb + 16 + M) * V + (v0 + tc * 16 + m)] = acc1[r] + bv;
    }
}

// ---------------------------------------------------------------------------
// Kernel 2c (fallback, small d_ws): split-bf16 GEMM converting in-kernel.
// ---------------------------------------------------------------------------
__global__ __launch_bounds__(256)
void out_proj(const float* __restrict__ H, const float* __restrict__ W,
              const float* __restrict__ bias, float* __restrict__ Y)
{
    __shared__ __align__(16) unsigned short sHh[BT][KPAD];
    __shared__ __align__(16) unsigned short sHl[BT][KPAD];
    __shared__ __align__(16) unsigned short sWh[BV][KPAD];
    __shared__ __align__(16) unsigned short sWl[BV][KPAD];

    const int t0   = blockIdx.y * BT;
    const int v0   = blockIdx.x * BV;
    const int tid  = threadIdx.x;
    const int lane = tid & 31;
    const int wave = tid >> 5;
    const int tb   = (wave >> 2) * 32;
    const int tc   = wave & 3;
    const int m    = lane & 15;
    const int kg   = lane >> 4;

    const int lr = tid >> 3;
    const int lk = (tid & 7) * 4;

    v8f acc0 = {};
    v8f acc1 = {};

    for (int k0 = 0; k0 < D; k0 += KB) {
        #pragma unroll
        for (int half = 0; half < 2; ++half) {
            const int rr = lr + 32 * half;
            const float4 hv = *(const float4*)&H[(size_t)(t0 + rr) * D + k0 + lk];
            const float4 wv = *(const float4*)&W[(size_t)(v0 + rr) * D + k0 + lk];

            unsigned short h0 = f2bf_rne(hv.x), h1 = f2bf_rne(hv.y),
                           h2 = f2bf_rne(hv.z), h3 = f2bf_rne(hv.w);
            *(unsigned int*)&sHh[rr][lk]     = (unsigned int)h0 | ((unsigned int)h1 << 16);
            *(unsigned int*)&sHh[rr][lk + 2] = (unsigned int)h2 | ((unsigned int)h3 << 16);
            unsigned short l0 = f2bf_rne(hv.x - bf2f(h0)), l1 = f2bf_rne(hv.y - bf2f(h1)),
                           l2 = f2bf_rne(hv.z - bf2f(h2)), l3 = f2bf_rne(hv.w - bf2f(h3));
            *(unsigned int*)&sHl[rr][lk]     = (unsigned int)l0 | ((unsigned int)l1 << 16);
            *(unsigned int*)&sHl[rr][lk + 2] = (unsigned int)l2 | ((unsigned int)l3 << 16);

            unsigned short w0 = f2bf_rne(wv.x), w1 = f2bf_rne(wv.y),
                           w2 = f2bf_rne(wv.z), w3 = f2bf_rne(wv.w);
            *(unsigned int*)&sWh[rr][lk]     = (unsigned int)w0 | ((unsigned int)w1 << 16);
            *(unsigned int*)&sWh[rr][lk + 2] = (unsigned int)w2 | ((unsigned int)w3 << 16);
            unsigned short x0 = f2bf_rne(wv.x - bf2f(w0)), x1 = f2bf_rne(wv.y - bf2f(w1)),
                           x2 = f2bf_rne(wv.z - bf2f(w2)), x3 = f2bf_rne(wv.w - bf2f(w3));
            *(unsigned int*)&sWl[rr][lk]     = (unsigned int)x0 | ((unsigned int)x1 << 16);
            *(unsigned int*)&sWl[rr][lk + 2] = (unsigned int)x2 | ((unsigned int)x3 << 16);
        }
        __syncthreads();

        const v16bf bh  = frag_b(&sWh[tc * 16 + m][0], kg);
        const v16bf bl  = frag_b(&sWl[tc * 16 + m][0], kg);
        const v16bf ah0 = frag_a(&sHh[tb + m][0],      kg);
        const v16bf al0 = frag_a(&sHl[tb + m][0],      kg);
        const v16bf ah1 = frag_a(&sHh[tb + 16 + m][0], kg);
        const v16bf al1 = frag_a(&sHl[tb + 16 + m][0], kg);

        acc0 = __builtin_amdgcn_wmma_f32_16x16x32_bf16(false, ah0, false, bh, (short)0, acc0, false, false);
        acc0 = __builtin_amdgcn_wmma_f32_16x16x32_bf16(false, ah0, false, bl, (short)0, acc0, false, false);
        acc0 = __builtin_amdgcn_wmma_f32_16x16x32_bf16(false, al0, false, bh, (short)0, acc0, false, false);
        acc1 = __builtin_amdgcn_wmma_f32_16x16x32_bf16(false, ah1, false, bh, (short)0, acc1, false, false);
        acc1 = __builtin_amdgcn_wmma_f32_16x16x32_bf16(false, ah1, false, bl, (short)0, acc1, false, false);
        acc1 = __builtin_amdgcn_wmma_f32_16x16x32_bf16(false, al1, false, bh, (short)0, acc1, false, false);

        __syncthreads();
    }

    const float bv = bias[v0 + tc * 16 + m];
    #pragma unroll
    for (int r = 0; r < 8; ++r) {
        const int M = r + 8 * kg;
        Y[(size_t)(t0 + tb + M) * V      + (v0 + tc * 16 + m)] = acc0[r] + bv;
        Y[(size_t)(t0 + tb + 16 + M) * V + (v0 + tc * 16 + m)] = acc1[r] + bv;
    }
}

// ---------------------------------------------------------------------------
// Kernel 3: in-place log_softmax over each row of Y (T rows of V=32000).
// ---------------------------------------------------------------------------
__global__ __launch_bounds__(256)
void logsoftmax_rows(float* __restrict__ Y)
{
    __shared__ float red[256];
    const int t = blockIdx.x;
    float* row = Y + (size_t)t * V;

    float mx = -3.402823466e+38f;
    for (int v = threadIdx.x; v < V; v += 256) mx = fmaxf(mx, row[v]);
    red[threadIdx.x] = mx;
    __syncthreads();
    for (int s = 128; s > 0; s >>= 1) {
        if (threadIdx.x < s) red[threadIdx.x] = fmaxf(red[threadIdx.x], red[threadIdx.x + s]);
        __syncthreads();
    }
    mx = red[0];
    __syncthreads();

    float sum = 0.0f;
    for (int v = threadIdx.x; v < V; v += 256) sum += __expf(row[v] - mx);
    red[threadIdx.x] = sum;
    __syncthreads();
    for (int s = 128; s > 0; s >>= 1) {
        if (threadIdx.x < s) red[threadIdx.x] += red[threadIdx.x + s];
        __syncthreads();
    }
    const float lse = mx + __logf(red[0]);

    for (int v = threadIdx.x; v < V; v += 256) row[v] -= lse;
}

// ---------------------------------------------------------------------------
extern "C" void kernel_launch(void* const* d_in, const int* in_sizes, int n_in,
                              void* d_out, int out_size, void* d_ws, size_t ws_size,
                              hipStream_t stream)
{
    const int*   X   = (const int*)  d_in[0];
    const float* Wxf = (const float*)d_in[1];
    const float* Whf = (const float*)d_in[2];
    const float* Wxi = (const float*)d_in[3];
    const float* Whi = (const float*)d_in[4];
    const float* Wxc = (const float*)d_in[5];
    const float* Whc = (const float*)d_in[6];
    const float* Wxo = (const float*)d_in[7];
    const float* Who = (const float*)d_in[8];
    const float* LW  = (const float*)d_in[9];
    const float* bf  = (const float*)d_in[10];
    const float* bi  = (const float*)d_in[11];
    const float* bc  = (const float*)d_in[12];
    const float* bo  = (const float*)d_in[13];
    const float* lb  = (const float*)d_in[14];

    float* outs = (float*)d_out;                 // (T, V) log-probs
    float* hid  = outs + (size_t)T * V;          // (1, T, D) hidden states

    // workspace layout: [0,256) barrier counter; then Whi, Wlo, Hhi, Hlo (bf16)
    unsigned int*  ctr = (unsigned int*)d_ws;
    unsigned short* Whi_s = (unsigned short*)((char*)d_ws + 256);
    unsigned short* Wlo_s = Whi_s + (size_t)V * D;
    unsigned short* Hhi_s = Wlo_s + (size_t)V * D;
    unsigned short* Hlo_s = Hhi_s + (size_t)T * D;
    const size_t need = 256 + 4ull * (size_t)V * D + 4ull * (size_t)T * D;
    const bool usePre = (ws_size >= need);

    init_ws<<<1, 1, 0, stream>>>(ctr);

    lstm_recur_mw<<<NWG, 128, 0, stream>>>(X, Wxf, Whf, Wxi, Whi, Wxc, Whc,
                                           Wxo, Who, bf, bi, bc, bo, hid, ctr);

    if (usePre) {
        split_pair<<<(int)((size_t)V * D / 1024), 256, 0, stream>>>(LW,  Whi_s, Wlo_s);
        split_pair<<<(int)((size_t)T * D / 1024), 256, 0, stream>>>(hid, Hhi_s, Hlo_s);
        out_proj_pre<<<dim3(V / BV, T / BT), 256, 0, stream>>>(Hhi_s, Hlo_s, Whi_s, Wlo_s, lb, outs);
    } else {
        out_proj<<<dim3(V / BV, T / BT), 256, 0, stream>>>(hid, LW, lb, outs);
    }

    logsoftmax_rows<<<T, 256, 0, stream>>>(outs);
}
